// CSNet_17420387352717
// MI455X (gfx1250) — compile-verified, matrix operand order checked by reading, not run on hardware
//
#include <hip/hip_runtime.h>
#include <hip/hip_bf16.h>
#include <math.h>

// ---------------- types ----------------
typedef __attribute__((ext_vector_type(16))) __bf16 v16bf;
typedef __attribute__((ext_vector_type(8)))  __bf16 v8bf;
typedef __attribute__((ext_vector_type(8)))  float  v8f;
typedef __attribute__((ext_vector_type(4)))  int    v4i;

#if defined(__HIP_DEVICE_COMPILE__)
typedef __attribute__((address_space(1))) v4i v4i_glob;   // global (AS1)
typedef __attribute__((address_space(3))) v4i v4i_lds;    // LDS (AS3)
#endif

union U16BF { v16bf v; v8bf h[2]; };

#if defined(__HIP_DEVICE_COMPILE__) && __has_builtin(__builtin_amdgcn_global_load_async_to_lds_b128)
#define HAVE_ASYNC_LDS 1
#else
#define HAVE_ASYNC_LDS 0
#endif

// ---------------- problem constants ----------------
#define BATCH   128
#define NCH     64
#define TLEN    1000
#define SMERGE  32
#define NFEAT   192            // 3 filters * 2 temp_exp * 32
#define SUMSP   1078
#define FCIN    18528          // 193*192/2
#define NCLASS  4
// BN scale 1/sqrt(1+1e-5)
#define BN_INV  0.9999950000374997f

// workspace layout (float offsets)
#define OFF_W     0            // 32*64
#define OFF_BEFF  2048         // 32
#define OFF_TFW   2080         // 3*64*64 folded temporal weights (zero padded)
#define OFF_TFB   14368        // 3*64 folded temporal biases
#define OFF_XS    14560        // 128*32*1000 f32
#define OFF_XT_F  (14560 + 4096000)   // bf16 region starts here (byte off 16,442,240; 16B aligned)
#define OFF_XB    16398560     // 128*192*192 f32 Gram matrices (lower triangle valid)

// ============================================================
// Kernel 0: fold spatial convs + merge conv + BN into W_eff (32x64), b_eff;
//           fold temporal conv weights with BN into tfw (3x64x64 padded), tfb.
// ============================================================
__global__ void k_prep(
    const float* __restrict__ sw0, const float* __restrict__ sb0,
    const float* __restrict__ sw1, const float* __restrict__ sb1,
    const float* __restrict__ sw2, const float* __restrict__ sb2,
    const float* __restrict__ sw3, const float* __restrict__ sb3,
    const float* __restrict__ sw4, const float* __restrict__ sb4,
    const float* __restrict__ mw,  const float* __restrict__ mb,
    const float* __restrict__ mg,  const float* __restrict__ mbt,
    const float* __restrict__ tw0, const float* __restrict__ tb0,
    const float* __restrict__ tg0, const float* __restrict__ tbt0,
    const float* __restrict__ tw1, const float* __restrict__ tb1,
    const float* __restrict__ tg1, const float* __restrict__ tbt1,
    const float* __restrict__ tw2, const float* __restrict__ tb2,
    const float* __restrict__ tg2, const float* __restrict__ tbt2,
    float* __restrict__ W, float* __restrict__ beff,
    float* __restrict__ tfw, float* __restrict__ tfb)
{
    const int tid = threadIdx.x;
    const int d0s[5] = {3, 4, 4, 7, 240};
    const int d1s[5] = {4, 8, 16, 32, 64};
    const float* sws[5] = {sw0, sw1, sw2, sw3, sw4};
    const float* sbs[5] = {sb0, sb1, sb2, sb3, sb4};

    // W_eff[m][c] = mg[m]*inv * sum_s mw[m][s] * sw_i[f][c-h]
    for (int p = tid; p < 32 * 64; p += 256) {
        int m = p >> 6, c = p & 63;
        float acc = 0.0f;
        int s = 0;
        for (int i = 0; i < 5; ++i) {
            int d0 = d0s[i], d1 = d1s[i], ho = NCH - d1 + 1;
            const float* swp = sws[i];
            for (int f = 0; f < d0; ++f)
                for (int h = 0; h < ho; ++h, ++s) {
                    int k = c - h;
                    if (k >= 0 && k < d1)
                        acc += mw[m * SUMSP + s] * swp[f * d1 + k];
                }
        }
        W[p] = mg[m] * BN_INV * acc;
    }

    // b_eff[m] = mg[m]*inv*(mb[m] + sum_s mw[m][s]*sb_i[f]) + mbt[m]
    if (tid < 32) {
        int m = tid;
        float bacc = mb[m];
        int s = 0;
        for (int i = 0; i < 5; ++i) {
            int d0 = d0s[i], d1 = d1s[i], ho = NCH - d1 + 1;
            const float* sbp = sbs[i];
            for (int f = 0; f < d0; ++f)
                for (int h = 0; h < ho; ++h, ++s)
                    bacc += mw[m * SUMSP + s] * sbp[f];
        }
        beff[m] = mg[m] * BN_INV * bacc + mbt[m];
    }

    // folded temporal weights/biases
    const int sizes[3] = {41, 51, 61};
    const float* tws[3] = {tw0, tw1, tw2};
    const float* tbs[3] = {tb0, tb1, tb2};
    const float* tgs[3] = {tg0, tg1, tg2};
    const float* tbts[3] = {tbt0, tbt1, tbt2};
    for (int p = tid; p < 3 * 64 * 64; p += 256) {
        int fi = p / 4096, j = (p >> 6) % 64, k = p & 63;
        int sz = sizes[fi];
        float v = 0.0f;
        if (k < sz) v = tws[fi][j * sz + k] * tgs[fi][j] * BN_INV;
        tfw[p] = v;
    }
    for (int p = tid; p < 3 * 64; p += 256) {
        int fi = p / 64, j = p & 63;
        tfb[p] = tgs[fi][j] * BN_INV * tbs[fi][j] + tbts[fi][j];
    }
}

// ============================================================
// Kernel 1: Xs[b][m][t] = W_eff x[b][:,t] + b_eff   (32x64 matvec per column)
// ============================================================
__global__ void __launch_bounds__(256) k_xs(
    const float* __restrict__ x, const float* __restrict__ W,
    const float* __restrict__ beff, float* __restrict__ Xs)
{
    __shared__ float Wsh[32 * 64];
    int tid = threadIdx.x;
    for (int i = tid; i < 2048; i += 256) Wsh[i] = W[i];
    __syncthreads();

    int flat = blockIdx.x * 256 + tid;         // 500 blocks * 256 = 128000
    int b = flat / TLEN, t = flat - b * TLEN;
    const float* xb = x + (size_t)b * NCH * TLEN + t;

    float acc[32];
#pragma unroll
    for (int m = 0; m < 32; ++m) acc[m] = beff[m];
    for (int c = 0; c < NCH; ++c) {
        float xv = xb[(size_t)c * TLEN];
#pragma unroll
        for (int m = 0; m < 32; ++m) acc[m] += Wsh[m * 64 + c] * xv;
    }
    float* out = Xs + (size_t)b * SMERGE * TLEN + t;
#pragma unroll
    for (int m = 0; m < 32; ++m) out[(size_t)m * TLEN] = acc[m];
}

// ============================================================
// Kernel 2: temporal grouped FIR convs (41/51/61 taps, same-pad) + BN,
//           output Xt in bf16. grid = (4 t-tiles, 32 groups, 128 batches)
// ============================================================
__global__ void __launch_bounds__(256) k_temporal(
    const float* __restrict__ Xs, const float* __restrict__ tfw,
    const float* __restrict__ tfb, __bf16* __restrict__ Xt)
{
    __shared__ float sxs[250 + 60];   // tile + max halo (30 each side)
    __shared__ float wsh[6 * 64];
    __shared__ float bsh[6];

    int tid = threadIdx.x;
    int t0  = blockIdx.x * 250;
    int g   = blockIdx.y;
    int b   = blockIdx.z;

    const float* xrow = Xs + ((size_t)b * SMERGE + g) * TLEN;
    for (int idx = tid; idx < 310; idx += 256) {
        int gt = t0 - 30 + idx;
        sxs[idx] = (gt >= 0 && gt < TLEN) ? xrow[gt] : 0.0f;
    }
    for (int idx = tid; idx < 384; idx += 256) {
        int row = idx >> 6, k = idx & 63;      // row = fi*2 + e
        int fi = row >> 1, e = row & 1;
        wsh[idx] = tfw[((size_t)fi * 64 + 2 * g + e) * 64 + k];
    }
    if (tid < 6) {
        int fi = tid >> 1, e = tid & 1;
        bsh[tid] = tfb[fi * 64 + 2 * g + e];
    }
    __syncthreads();

    if (tid < 250) {
        int t = t0 + tid;
        const int sizes[3] = {41, 51, 61};
        const int pads[3]  = {20, 25, 30};
#pragma unroll
        for (int fi = 0; fi < 3; ++fi) {
            int sz = sizes[fi], pad = pads[fi];
#pragma unroll
            for (int e = 0; e < 2; ++e) {
                int row = fi * 2 + e;
                float acc = bsh[row];
                for (int dt = 0; dt < sz; ++dt)
                    acc += wsh[row * 64 + dt] * sxs[tid + 30 + dt - pad];
                size_t oidx = ((size_t)b * NFEAT + fi * 64 + 2 * g + e) * TLEN + t;
                Xt[oidx] = (__bf16)acc;
            }
        }
    }
}

// ============================================================
// Kernel 3: Gram matrix Xb = Xt Xt^T / (T-1) via v_wmma_f32_16x16x32_bf16.
// One workgroup (16 waves) per batch; 78 lower-triangle 16x16 tiles.
// K staged through LDS in 128-element chunks (padded row stride 136 bf16).
// Double-buffered: next chunk is copied with GLOBAL_LOAD_ASYNC_TO_LDS_B128
// (ASYNCcnt) while the current chunk feeds the WMMAs; falls back to a
// register-staged software pipeline if the async builtin is unavailable.
// ============================================================
__global__ void __launch_bounds__(512) k_gram(
    const __bf16* __restrict__ Xt, float* __restrict__ Xb)
{
#if HAVE_ASYNC_LDS
    __shared__ __bf16 lds0[192 * 136];         // 2 x 51 KB double buffer
    __shared__ __bf16 lds1[192 * 136];
#else
    __shared__ __bf16 lds0[192 * 136];
#endif

    int tid  = threadIdx.x;
    int b    = blockIdx.x;
    int wave = tid >> 5;
    int lane = tid & 31;
    int m    = lane & 15;
    int hi   = lane >> 4;

    v8f acc[5] = {};
    int tis[5] = {0, 0, 0, 0, 0}, tjs[5] = {0, 0, 0, 0, 0};
#pragma unroll
    for (int u = 0; u < 5; ++u) {
        int tIdx = wave + u * 16;
        if (tIdx < 78) {
            int ti = (int)((sqrtf(8.0f * (float)tIdx + 1.0f) - 1.0f) * 0.5f);
            while ((ti + 1) * (ti + 2) / 2 <= tIdx) ++ti;
            while (ti * (ti + 1) / 2 > tIdx) --ti;
            tis[u] = ti;
            tjs[u] = tIdx - ti * (ti + 1) / 2;
        }
    }

    const __bf16* xtb = Xt + (size_t)b * NFEAT * TLEN;

#if HAVE_ASYNC_LDS
    // ---- async double-buffered pipeline ----
    // issue chunk 0 -> lds0
    {
        for (int q = tid; q < 192 * 16; q += 512) {
            int r = q >> 4, sub = q & 15;
            int t = 0 * 128 + sub * 8;                   // always < TLEN for kc=0
            const __bf16* g = xtb + (size_t)r * TLEN + t;
            __bf16* l = &lds0[r * 136 + sub * 8];
            __builtin_amdgcn_global_load_async_to_lds_b128(
                (v4i_glob*)(void*)g, (v4i_lds*)(void*)l, 0, 0);
        }
    }

    for (int kc = 0; kc < 8; ++kc) {
        __bf16* cur = (kc & 1) ? lds1 : lds0;
        __bf16* nxt = (kc & 1) ? lds0 : lds1;

#if __has_builtin(__builtin_amdgcn_s_wait_asynccnt)
        __builtin_amdgcn_s_wait_asynccnt(0);
#else
        asm volatile("s_wait_asynccnt 0" ::: "memory");
#endif
        __syncthreads();                       // chunk kc landed for all waves

        if (kc < 7) {                          // overlap copy of kc+1 with compute
            int kn = kc + 1;
            for (int q = tid; q < 192 * 16; q += 512) {
                int r = q >> 4, sub = q & 15;
                int t = kn * 128 + sub * 8;
                if (t < TLEN) {                // masked lanes skip (tail)
                    const __bf16* g = xtb + (size_t)r * TLEN + t;
                    __bf16* l = &nxt[r * 136 + sub * 8];
                    __builtin_amdgcn_global_load_async_to_lds_b128(
                        (v4i_glob*)(void*)g, (v4i_lds*)(void*)l, 0, 0);
                }
            }
        }
        if (kc == 7) {                         // zero stale tail cols 104..127
            for (int q = tid; q < 192 * 12; q += 512) {
                int r = q / 12, c = 52 + (q - r * 12);
                reinterpret_cast<unsigned int*>(&cur[r * 136])[c] = 0u;
            }
            __syncthreads();
        }

#pragma unroll
        for (int u = 0; u < 5; ++u) {
            int tIdx = wave + u * 16;
            if (tIdx >= 78) continue;
            int ra = tis[u] * 16 + m;
            int rb = tjs[u] * 16 + m;
#pragma unroll
            for (int kk = 0; kk < 4; ++kk) {
                int koff = kk * 32 + hi * 8;
                U16BF ua, ub;
                ua.h[0] = *reinterpret_cast<const v8bf*>(&cur[ra * 136 + koff]);
                ua.h[1] = *reinterpret_cast<const v8bf*>(&cur[ra * 136 + koff + 16]);
                ub.h[0] = *reinterpret_cast<const v8bf*>(&cur[rb * 136 + koff]);
                ub.h[1] = *reinterpret_cast<const v8bf*>(&cur[rb * 136 + koff + 16]);
                acc[u] = __builtin_amdgcn_wmma_f32_16x16x32_bf16(
                    false, ua.v, false, ub.v, (short)0, acc[u], false, false);
            }
        }
    }
#else
    // ---- fallback: register-staged software pipeline (single LDS buffer) ----
    uint4 regs[6];
    {
        // preload chunk 0
#pragma unroll
        for (int n = 0; n < 6; ++n) {
            int q = tid + n * 512;
            int r = q >> 4, sub = q & 15;
            int t = sub * 8;
            regs[n] = *reinterpret_cast<const uint4*>(xtb + (size_t)r * TLEN + t);
        }
    }
    for (int kc = 0; kc < 8; ++kc) {
        __syncthreads();                       // previous chunk's readers done
#pragma unroll
        for (int n = 0; n < 6; ++n) {
            int q = tid + n * 512;
            int r = q >> 4, sub = q & 15;
            *reinterpret_cast<uint4*>(&lds0[r * 136 + sub * 8]) = regs[n];
        }
        __syncthreads();
        if (kc < 7) {                          // issue next chunk's loads early
            int kn = kc + 1;
#pragma unroll
            for (int n = 0; n < 6; ++n) {
                int q = tid + n * 512;
                int r = q >> 4, sub = q & 15;
                int t = kn * 128 + sub * 8;
                uint4 v = {0u, 0u, 0u, 0u};
                if (t < TLEN)
                    v = *reinterpret_cast<const uint4*>(xtb + (size_t)r * TLEN + t);
                regs[n] = v;
            }
        }
#pragma unroll
        for (int u = 0; u < 5; ++u) {
            int tIdx = wave + u * 16;
            if (tIdx >= 78) continue;
            int ra = tis[u] * 16 + m;
            int rb = tjs[u] * 16 + m;
#pragma unroll
            for (int kk = 0; kk < 4; ++kk) {
                int koff = kk * 32 + hi * 8;
                U16BF ua, ub;
                ua.h[0] = *reinterpret_cast<const v8bf*>(&lds0[ra * 136 + koff]);
                ua.h[1] = *reinterpret_cast<const v8bf*>(&lds0[ra * 136 + koff + 16]);
                ub.h[0] = *reinterpret_cast<const v8bf*>(&lds0[rb * 136 + koff]);
                ub.h[1] = *reinterpret_cast<const v8bf*>(&lds0[rb * 136 + koff + 16]);
                acc[u] = __builtin_amdgcn_wmma_f32_16x16x32_bf16(
                    false, ua.v, false, ub.v, (short)0, acc[u], false, false);
            }
        }
    }
#endif

    const float s = 1.0f / (float)(TLEN - 1);
    float* Gb = Xb + (size_t)b * NFEAT * NFEAT;
#pragma unroll
    for (int u = 0; u < 5; ++u) {
        int tIdx = wave + u * 16;
        if (tIdx >= 78) continue;
#pragma unroll
        for (int r = 0; r < 8; ++r) {
            int mm  = r + 8 * hi;              // C/D layout: vgpr r, lane-half selects M
            int row = tis[u] * 16 + mm;
            int col = tjs[u] * 16 + (lane & 15);
            Gb[(size_t)row * NFEAT + col] = acc[u][r] * s;
        }
    }
}

// ============================================================
// Kernel 4: per-batch Cholesky (192x192 in LDS), fused log-diag + tril
// tangent gather + FC (4 classes) with LDS reduction.
// ============================================================
__global__ void __launch_bounds__(256) k_chol_fc(
    const float* __restrict__ Xb, const float* __restrict__ fcw,
    const float* __restrict__ fcb, float* __restrict__ out)
{
    __shared__ float A[192 * 193];             // 148 KB, padded stride
    __shared__ float red[4 * 256];
    __shared__ float s_inv;

    int tid = threadIdx.x;
    int b   = blockIdx.x;
    const float* G = Xb + (size_t)b * NFEAT * NFEAT;

    for (int p = tid; p < NFEAT * NFEAT; p += 256) {
        int i = p / NFEAT, j = p - i * NFEAT;
        if (j <= i) A[i * 193 + j] = G[p];
    }
    __syncthreads();

    // right-looking Cholesky
    for (int j = 0; j < NFEAT; ++j) {
        if (tid == 0) {
            float d = sqrtf(A[j * 193 + j]);
            A[j * 193 + j] = d;
            s_inv = 1.0f / d;
        }
        __syncthreads();
        float inv = s_inv;
        for (int i = j + 1 + tid; i < NFEAT; i += 256) A[i * 193 + j] *= inv;
        __syncthreads();
        for (int i = j + 1 + tid; i < NFEAT; i += 256) {
            float lij = A[i * 193 + j];
            for (int k = j + 1; k <= i; ++k)
                A[i * 193 + k] -= lij * A[k * 193 + j];
        }
        __syncthreads();
    }

    // fused tangent + FC
    float a0 = 0.f, a1 = 0.f, a2 = 0.f, a3 = 0.f;
    for (int f = tid; f < NFEAT; f += 256) {
        float lg = logf(A[f * 193 + f]);
        a0 += lg * fcw[0 * FCIN + f];
        a1 += lg * fcw[1 * FCIN + f];
        a2 += lg * fcw[2 * FCIN + f];
        a3 += lg * fcw[3 * FCIN + f];
    }
    for (int p = tid; p < (NFEAT * (NFEAT - 1)) / 2; p += 256) {
        int i = (int)((1.0f + sqrtf(1.0f + 8.0f * (float)p)) * 0.5f);
        while (i * (i - 1) / 2 > p) --i;
        while (i * (i + 1) / 2 <= p) ++i;
        int jj  = p - i * (i - 1) / 2;
        float v = A[i * 193 + jj];
        int col = NFEAT + p;
        a0 += v * fcw[0 * FCIN + col];
        a1 += v * fcw[1 * FCIN + col];
        a2 += v * fcw[2 * FCIN + col];
        a3 += v * fcw[3 * FCIN + col];
    }
    red[tid] = a0; red[256 + tid] = a1; red[512 + tid] = a2; red[768 + tid] = a3;
    __syncthreads();
    if (tid < 4) {
        float s = 0.f;
        for (int q = 0; q < 256; ++q) s += red[tid * 256 + q];
        out[b * NCLASS + tid] = s + fcb[tid];
    }
}

// ============================================================
extern "C" void kernel_launch(void* const* d_in, const int* in_sizes, int n_in,
                              void* d_out, int out_size, void* d_ws, size_t ws_size,
                              hipStream_t stream)
{
    const float* x   = (const float*)d_in[0];
    const float* sw0 = (const float*)d_in[1];  const float* sb0 = (const float*)d_in[2];
    const float* sw1 = (const float*)d_in[3];  const float* sb1 = (const float*)d_in[4];
    const float* sw2 = (const float*)d_in[5];  const float* sb2 = (const float*)d_in[6];
    const float* sw3 = (const float*)d_in[7];  const float* sb3 = (const float*)d_in[8];
    const float* sw4 = (const float*)d_in[9];  const float* sb4 = (const float*)d_in[10];
    const float* mw  = (const float*)d_in[11]; const float* mb  = (const float*)d_in[12];
    const float* mg  = (const float*)d_in[13]; const float* mbt = (const float*)d_in[14];
    const float* tw0 = (const float*)d_in[15]; const float* tb0 = (const float*)d_in[16];
    const float* tg0 = (const float*)d_in[17]; const float* tbt0= (const float*)d_in[18];
    const float* tw1 = (const float*)d_in[19]; const float* tb1 = (const float*)d_in[20];
    const float* tg1 = (const float*)d_in[21]; const float* tbt1= (const float*)d_in[22];
    const float* tw2 = (const float*)d_in[23]; const float* tb2 = (const float*)d_in[24];
    const float* tg2 = (const float*)d_in[25]; const float* tbt2= (const float*)d_in[26];
    const float* fcw = (const float*)d_in[27]; const float* fcb = (const float*)d_in[28];

    float* ws   = (float*)d_ws;
    float* W    = ws + OFF_W;
    float* beff = ws + OFF_BEFF;
    float* tfw  = ws + OFF_TFW;
    float* tfb  = ws + OFF_TFB;
    float* Xs   = ws + OFF_XS;
    __bf16* Xt  = (__bf16*)(ws + OFF_XT_F);
    float* Xb   = ws + OFF_XB;
    float* out  = (float*)d_out;

    k_prep<<<1, 256, 0, stream>>>(sw0, sb0, sw1, sb1, sw2, sb2, sw3, sb3, sw4, sb4,
                                  mw, mb, mg, mbt,
                                  tw0, tb0, tg0, tbt0,
                                  tw1, tb1, tg1, tbt1,
                                  tw2, tb2, tg2, tbt2,
                                  W, beff, tfw, tfb);

    k_xs<<<(BATCH * TLEN) / 256, 256, 0, stream>>>(x, W, beff, Xs);

    k_temporal<<<dim3(4, SMERGE, BATCH), 256, 0, stream>>>(Xs, tfw, tfb, Xt);

    k_gram<<<BATCH, 512, 0, stream>>>(Xt, Xb);

    k_chol_fc<<<BATCH, 256, 0, stream>>>(Xb, fcw, fcb, out);
}
